// PerformerBaseline_85194971283979
// MI455X (gfx1250) — matrix operand order repaired and checked by reading, not compiled, not run on hardware
//
#include <hip/hip_runtime.h>
#include <math.h>

typedef _Float16 v16h __attribute__((ext_vector_type(16)));
typedef _Float16 v8h  __attribute__((ext_vector_type(8)));
typedef float    v8f  __attribute__((ext_vector_type(8)));

union h16u { v16h v; v8h h8[2]; };

// Activation codes: 0 = none, 1 = elu(x)+1, 2 = gelu(tanh approx)
// Branch-free: tanh(t) = 1 - 2/(exp(2t)+1) via hardware v_exp_f32.
__device__ __forceinline__ float apply_act(float x, int act) {
  if (act == 1) return x > 0.f ? (x + 1.f) : __expf(x);
  if (act == 2) {
    float t = 0.7978845608f * (x + 0.044715f * x * x * x);
    float e = __expf(2.f * t);
    float th = 1.f - 2.f / (e + 1.f);
    return 0.5f * x * (1.f + th);
  }
  return x;
}

// ---------------------------------------------------------------------------
// f16-operand WMMA GEMM with async-LDS double buffering.
//   C[z] = act( op(A[z]) @ B[z] + bias ) (+ resid[z])
//   A: M x K f16 row-major, or (TRANSA) K x M f16 (computes A^T @ B)
//   B: K x N f16 row-major
//   Outputs: Cf (fp32, optional) and Ch (f16, optional) — same values.
// REQUIREMENTS: M % 64 == 0, N % BN == 0, K % 32 == 0 (guaranteed by callers).
// Block: 256 threads = 8 waves, tile 64 x BN, K-step 32, ping-pong LDS.
//   NT A-tile: staged by GLOBAL_LOAD_ASYNC_TO_LDS_B128 (1 per thread/tile),
//              overlapped with WMMA, fenced by s_wait_asynccnt + barrier.
//   B-tile (and TRANSA A-tile): registers -> transposed LDS stores.
// ---------------------------------------------------------------------------
template <int BN, bool TRANSA>
__global__ __launch_bounds__(256)
void perf_wmma_gemm(const _Float16* __restrict__ A, const _Float16* __restrict__ Bw,
                    const float* __restrict__ bias, const float* __restrict__ resid,
                    float* __restrict__ Cf, _Float16* __restrict__ Ch,
                    int M, int N, int K, int act,
                    long long sAz, long long sBz, long long sCz) {
  constexpr int WCW = (BN == 128) ? 4 : 2;   // waves across N
  constexpr int WRW = 8 / WCW;               // waves across M
  constexpr int TR  = 64 / (WRW * 16);       // M tiles per wave
  constexpr int TC  = BN / (WCW * 16);       // N tiles per wave
  constexpr int B8  = (32 * BN / 8) / 256;   // v8h B loads per thread per tile

  __shared__ _Float16 ldsA[2][64 * 40]  __attribute__((aligned(16)));
  __shared__ _Float16 ldsB[2][BN * 40]  __attribute__((aligned(16)));

  const int tid  = threadIdx.x;
  const int lane = tid & 31;
  const int wave = tid >> 5;
  const int wr   = wave / WCW;
  const int wc   = wave % WCW;
  const int g    = lane >> 4;
  const int ml   = lane & 15;

  const int row0 = blockIdx.y * 64;
  const int col0 = blockIdx.x * BN;
  const long long z = blockIdx.z;

  A  += z * sAz;
  Bw += z * sBz;
  if (Cf) Cf += z * sCz;
  if (Ch) Ch += z * sCz;
  const float* R = resid ? (resid + z * sCz) : nullptr;

  v8f acc[TR][TC];
#pragma unroll
  for (int tr = 0; tr < TR; ++tr)
#pragma unroll
    for (int tc = 0; tc < TC; ++tc) acc[tr][tc] = {};

  v8h br[B8];   // B tile staging registers
  v8h atr;      // A tile staging register (TRANSA only)

  // ---- NT A tile: async global->LDS copy, 16B per lane, whole tile/block ----
  auto async_copy_A = [&](int k0, int buf) {
    int m  = tid >> 2;                 // 64 rows, 4 chunks of 8 halfs per row
    int kc = (tid & 3) << 3;
    unsigned lo = (unsigned)(size_t)&ldsA[buf][m * 40 + kc];
    const void* gp = &A[(size_t)(row0 + m) * (size_t)K + k0 + kc];
    asm volatile("global_load_async_to_lds_b128 %0, %1, off"
                 :: "v"(lo), "v"(gp) : "memory");
  };

  // ---- TRANSA A tile: registers (global contiguous in m) ----
  auto load_A_regs = [&](int k0) {
    int kk = tid >> 3, mc = (tid & 7) << 3;     // 32 k-rows x 8 chunks
    atr = *(const v8h*)&A[(size_t)(k0 + kk) * (size_t)M + row0 + mc];
  };
  auto store_A_regs = [&](int buf) {
    int kk = tid >> 3, mc = (tid & 7) << 3;
#pragma unroll
    for (int j = 0; j < 8; ++j) ldsA[buf][(mc + j) * 40 + kk] = atr[j];
  };

  // ---- B tile: registers, transposed LDS stores ----
  auto load_B_regs = [&](int k0) {
#pragma unroll
    for (int i = 0; i < B8; ++i) {
      int idx = i * 256 + tid;
      int kk = idx / (BN / 8), nc = (idx % (BN / 8)) << 3;
      br[i] = *(const v8h*)&Bw[(size_t)(k0 + kk) * (size_t)N + col0 + nc];
    }
  };
  auto store_B_regs = [&](int buf) {
#pragma unroll
    for (int i = 0; i < B8; ++i) {
      int idx = i * 256 + tid;
      int kk = idx / (BN / 8), nc = (idx % (BN / 8)) << 3;
#pragma unroll
      for (int j = 0; j < 8; ++j) ldsB[buf][(nc + j) * 40 + kk] = br[i][j];
    }
  };

  auto prefetch_tile = [&](int k0) {
    if (!TRANSA) {
      int m = tid >> 2;
      __builtin_prefetch(&A[(size_t)(row0 + m) * (size_t)K + k0 + ((tid & 3) << 3)], 0, 1);
    }
    int kk = tid / (BN / 32);
    __builtin_prefetch(&Bw[(size_t)(k0 + kk) * (size_t)N + col0 + ((tid % (BN / 32)) << 5)], 0, 1);
  };

  // ---- prologue: stage tile 0 into buffer 0 ----
  if constexpr (!TRANSA) {
    async_copy_A(0, 0);
  } else {
    load_A_regs(0);
    store_A_regs(0);
  }
  load_B_regs(0);
  store_B_regs(0);

  int cur = 0;
  for (int k0 = 0; k0 < K; k0 += 32) {
    if constexpr (!TRANSA)
      asm volatile("s_wait_asynccnt 0" ::: "memory");   // my async copies landed
    __syncthreads();                                    // tile(cur) visible to all

    const bool has_next = (k0 + 32) < K;
    if (has_next) {
      // stage next tile into cur^1 (safe: all waves finished reading it)
      if constexpr (!TRANSA) async_copy_A(k0 + 32, cur ^ 1);
      else                   load_A_regs(k0 + 32);
      load_B_regs(k0 + 32);
      if (k0 + 64 < K) prefetch_tile(k0 + 64);
    }

    // ---- fragments (contiguous k = 16g + s in A and B) + WMMA on tile(cur) ----
    v16h af[TR], bf[TC];
#pragma unroll
    for (int tr = 0; tr < TR; ++tr) {
      int arow = wr * (TR * 16) + tr * 16 + ml;
      h16u u;
      u.h8[0] = *(const v8h*)&ldsA[cur][arow * 40 + 16 * g];      // K = 16g..+7
      u.h8[1] = *(const v8h*)&ldsA[cur][arow * 40 + 16 * g + 8];  // K = 16g+8..+15
      af[tr] = u.v;
    }
#pragma unroll
    for (int tc = 0; tc < TC; ++tc) {
      int bcol = wc * (TC * 16) + tc * 16 + ml;
      h16u u;
      u.h8[0] = *(const v8h*)&ldsB[cur][bcol * 40 + 16 * g];      // K = 16g..+7
      u.h8[1] = *(const v8h*)&ldsB[cur][bcol * 40 + 16 * g + 8];  // K = 16g+8..+15
      bf[tc] = u.v;
    }
#pragma unroll
    for (int tr = 0; tr < TR; ++tr)
#pragma unroll
      for (int tc = 0; tc < TC; ++tc)
        acc[tr][tc] = __builtin_amdgcn_wmma_f32_16x16x32_f16(
            false, af[tr], false, bf[tc], (short)0, acc[tr][tc], false, false);

    if (has_next) {
      store_B_regs(cur ^ 1);                      // after compute; visible next iter
      if constexpr (TRANSA) store_A_regs(cur ^ 1);
    }
    cur ^= 1;
  }

  // ---- epilogue: bias + activation + residual; dual fp32/f16 outputs ----
#pragma unroll
  for (int tr = 0; tr < TR; ++tr) {
#pragma unroll
    for (int tc = 0; tc < TC; ++tc) {
      int n = col0 + wc * (TC * 16) + tc * 16 + ml;
      float bv = bias ? bias[n] : 0.f;
#pragma unroll
      for (int r = 0; r < 8; ++r) {
        int m = row0 + wr * (TR * 16) + tr * 16 + 8 * g + r;  // C/D: M = r + 8g
        float v = acc[tr][tc][r] + bv;
        v = apply_act(v, act);
        if (R) v += R[(size_t)m * (size_t)N + n];
        if (Cf) Cf[(size_t)m * (size_t)N + n] = v;
        if (Ch) Ch[(size_t)m * (size_t)N + n] = (_Float16)v;
      }
    }
  }
}

// ---------------------------------------------------------------------------
// fp32 -> f16 bulk convert (weights, once per launch)
// ---------------------------------------------------------------------------
__global__ void perf_f32tof16(const float* __restrict__ src, _Float16* __restrict__ dst,
                              long long n) {
  long long i = (long long)blockIdx.x * 256 + threadIdx.x;
  long long stride = (long long)gridDim.x * 256;
  for (; i < n; i += stride) dst[i] = (_Float16)src[i];
}

// ---------------------------------------------------------------------------
// h[b,s,:] = emb[x[b,s],:] + pos[s,:]  (fp32 + f16 outputs)
// ---------------------------------------------------------------------------
__global__ void perf_embed(const int* __restrict__ x, const float* __restrict__ emb,
                           const float* __restrict__ pos, float* __restrict__ h,
                           _Float16* __restrict__ h16, int S, int E) {
  int bs = blockIdx.x;
  int s  = bs % S;
  int v  = x[bs];
  const float* er = emb + (size_t)v * E;
  const float* pr = pos + (size_t)s * E;
  float*     hr  = h   + (size_t)bs * E;
  _Float16*  hr6 = h16 + (size_t)bs * E;
  for (int e = threadIdx.x; e < E; e += blockDim.x) {
    float t = er[e] + pr[e];
    hr[e] = t;
    hr6[e] = (_Float16)t;
  }
}

// ---------------------------------------------------------------------------
// z[b,p] = sum_s phik[b,s,p]  (f16 input, fp32 accumulate)
// ---------------------------------------------------------------------------
__global__ void perf_zreduce(const _Float16* __restrict__ phik, float* __restrict__ z,
                             int S, int P) {
  __shared__ float red[256];
  int b = blockIdx.x / P;
  int p = blockIdx.x % P;
  const _Float16* base = phik + (size_t)b * S * P + p;
  float acc = 0.f;
  for (int s = threadIdx.x; s < S; s += 256) acc += (float)base[(size_t)s * P];
  red[threadIdx.x] = acc;
  __syncthreads();
  for (int off = 128; off > 0; off >>= 1) {
    if (threadIdx.x < off) red[threadIdx.x] += red[threadIdx.x + off];
    __syncthreads();
  }
  if (threadIdx.x == 0) z[blockIdx.x] = red[0];
}

// ---------------------------------------------------------------------------
// den[bs] = dot(phiq[bs,:], z[b,:]) + eps ; attn16[bs,:] = num[bs,:] / den
// ---------------------------------------------------------------------------
__global__ void perf_denscale(const _Float16* __restrict__ phiq, const float* __restrict__ z,
                              const float* __restrict__ num, _Float16* __restrict__ attn16,
                              int S, int P, int E) {
  __shared__ float red[64];
  __shared__ float dinv;
  int bs = blockIdx.x;
  int b  = bs / S;
  red[threadIdx.x] = (float)phiq[(size_t)bs * P + threadIdx.x] * z[(size_t)b * P + threadIdx.x];
  __syncthreads();
  for (int off = 32; off > 0; off >>= 1) {
    if (threadIdx.x < off) red[threadIdx.x] += red[threadIdx.x + off];
    __syncthreads();
  }
  if (threadIdx.x == 0) dinv = 1.f / (red[0] + 1e-6f);
  __syncthreads();
  const float* nrow = num + (size_t)bs * E;
  _Float16* arow = attn16 + (size_t)bs * E;
  for (int e = threadIdx.x; e < E; e += 64) arow[e] = (_Float16)(nrow[e] * dinv);
}

// ---------------------------------------------------------------------------
// Row layernorm: fp32 in, f16 out
// ---------------------------------------------------------------------------
__global__ void perf_layernorm(const float* __restrict__ in, _Float16* __restrict__ out16,
                               const float* __restrict__ g, const float* __restrict__ b,
                               int E) {
  __shared__ float red[256];
  const float* row = in + (size_t)blockIdx.x * E;
  _Float16* orow = out16 + (size_t)blockIdx.x * E;
  float s = 0.f;
  for (int e = threadIdx.x; e < E; e += 256) s += row[e];
  red[threadIdx.x] = s;
  __syncthreads();
  for (int off = 128; off > 0; off >>= 1) {
    if (threadIdx.x < off) red[threadIdx.x] += red[threadIdx.x + off];
    __syncthreads();
  }
  float mu = red[0] / (float)E;
  __syncthreads();
  float v = 0.f;
  for (int e = threadIdx.x; e < E; e += 256) {
    float d = row[e] - mu;
    v += d * d;
  }
  red[threadIdx.x] = v;
  __syncthreads();
  for (int off = 128; off > 0; off >>= 1) {
    if (threadIdx.x < off) red[threadIdx.x] += red[threadIdx.x + off];
    __syncthreads();
  }
  float rs = rsqrtf(red[0] / (float)E + 1e-5f);
  __syncthreads();
  for (int e = threadIdx.x; e < E; e += 256)
    orow[e] = (_Float16)((row[e] - mu) * rs * g[e] + b[e]);
}

// ---------------------------------------------------------------------------
// Head: out[b,:] = relu(h[b,0,:] @ Wh1 + bh1) @ Wh2 + bh2  (fp32)
// ---------------------------------------------------------------------------
__global__ void perf_head(const float* __restrict__ h, const float* __restrict__ Wh1,
                          const float* __restrict__ bh1, const float* __restrict__ Wh2,
                          const float* __restrict__ bh2, float* __restrict__ out,
                          int S, int E, int H, int C) {
  __shared__ float pooled[512];
  __shared__ float hid[256];
  int b = blockIdx.x;
  for (int e = threadIdx.x; e < E; e += 256) pooled[e] = h[(size_t)b * S * E + e];
  __syncthreads();
  if (threadIdx.x < H) {
    float a = bh1[threadIdx.x];
    for (int e = 0; e < E; ++e) a += pooled[e] * Wh1[(size_t)e * H + threadIdx.x];
    hid[threadIdx.x] = fmaxf(a, 0.f);
  }
  __syncthreads();
  if (threadIdx.x < C) {
    float o = bh2[threadIdx.x];
    for (int j = 0; j < H; ++j) o += hid[j] * Wh2[(size_t)j * C + threadIdx.x];
    out[(size_t)b * C + threadIdx.x] = o;
  }
}

// ---------------------------------------------------------------------------
// Host orchestration
// ---------------------------------------------------------------------------
template <int BN, bool TRANSA>
static inline void launch_gemm(const _Float16* A, const _Float16* Bw, const float* bias,
                               const float* resid, float* Cf, _Float16* Ch,
                               int M, int N, int K, int act,
                               long long sAz, long long sBz, long long sCz, int Z,
                               hipStream_t stream) {
  dim3 grid((N + BN - 1) / BN, (M + 63) / 64, Z);
  perf_wmma_gemm<BN, TRANSA><<<grid, dim3(256), 0, stream>>>(
      A, Bw, bias, resid, Cf, Ch, M, N, K, act, sAz, sBz, sCz);
}

static inline void cvt16(const float* s, _Float16* d, long long n, hipStream_t st) {
  unsigned blocks = (unsigned)((n + 255) / 256);
  if (blocks > 4096) blocks = 4096;
  perf_f32tof16<<<dim3(blocks), dim3(256), 0, st>>>(s, d, n);
}

extern "C" void kernel_launch(void* const* d_in, const int* in_sizes, int n_in,
                              void* d_out, int out_size, void* d_ws, size_t ws_size,
                              hipStream_t stream) {
  constexpr int B = 8, S = 4096, E = 512, P = 64, L = 4, H = 256, C = 2;
  constexpr long long BS = (long long)B * S;
  constexpr int F = 4 * E;           // 2048
  constexpr int CHUNK = 4096;        // FFN row chunk

  const int*   x    = (const int*)d_in[0];
  const float* emb  = (const float*)d_in[1];
  const float* pos  = (const float*)d_in[2];
  const float* Wq   = (const float*)d_in[3];
  const float* Wk   = (const float*)d_in[4];
  const float* Wv   = (const float*)d_in[5];
  const float* Wo   = (const float*)d_in[6];
  const float* ln_g = (const float*)d_in[7];
  const float* ln_b = (const float*)d_in[8];
  const float* W1   = (const float*)d_in[9];
  const float* b1   = (const float*)d_in[10];
  const float* W2   = (const float*)d_in[11];
  const float* b2   = (const float*)d_in[12];
  const float* Wh1  = (const float*)d_in[13];
  const float* bh1  = (const float*)d_in[14];
  const float* Wh2  = (const float*)d_in[15];
  const float* bh2  = (const float*)d_in[16];
  float* out = (float*)d_out;

  char* ws = (char*)d_ws;
  auto allocF = [&](size_t n) { float* p = (float*)ws; ws += n * sizeof(float); return p; };
  auto allocH = [&](size_t n) { _Float16* p = (_Float16*)ws; ws += n * sizeof(_Float16); return p; };

  // fp32 buffers
  float* hFA = allocF((size_t)BS * E);          // h ping (residual source)
  float* hFB = allocF((size_t)BS * E);          // h pong
  float* tF  = allocF((size_t)BS * E);          // num, then a
  float* zb  = allocF((size_t)B * P);
  // f16 buffers
  _Float16* h16A   = allocH((size_t)BS * E);
  _Float16* h16B   = allocH((size_t)BS * E);
  _Float16* s16    = allocH((size_t)BS * E);    // v16 -> attn16 -> n16 (sequential reuse)
  _Float16* phiq16 = allocH((size_t)BS * P);
  _Float16* phik16 = allocH((size_t)BS * P);
  _Float16* kv16   = allocH((size_t)B * P * E);
  _Float16* m16    = allocH((size_t)CHUNK * F);
  // f16 weights
  _Float16* wq16 = allocH((size_t)L * E * P);
  _Float16* wk16 = allocH((size_t)L * E * P);
  _Float16* wv16 = allocH((size_t)L * E * E);
  _Float16* wo16 = allocH((size_t)L * E * E);
  _Float16* w116 = allocH((size_t)L * E * F);
  _Float16* w216 = allocH((size_t)L * F * E);

  // ---- one-time weight conversion ----
  cvt16(Wq, wq16, (long long)L * E * P, stream);
  cvt16(Wk, wk16, (long long)L * E * P, stream);
  cvt16(Wv, wv16, (long long)L * E * E, stream);
  cvt16(Wo, wo16, (long long)L * E * E, stream);
  cvt16(W1, w116, (long long)L * E * F, stream);
  cvt16(W2, w216, (long long)L * F * E, stream);

  perf_embed<<<dim3((unsigned)BS), dim3(128), 0, stream>>>(x, emb, pos, hFA, h16A, S, E);

  float*    hF  = hFA; float*    hFalt  = hFB;
  _Float16* h16 = h16A; _Float16* h16alt = h16B;

  for (int l = 0; l < L; ++l) {
    const _Float16* wq = wq16 + (size_t)l * E * P;
    const _Float16* wk = wk16 + (size_t)l * E * P;
    const _Float16* wv = wv16 + (size_t)l * E * E;
    const _Float16* wo = wo16 + (size_t)l * E * E;
    const _Float16* w1 = w116 + (size_t)l * E * F;
    const _Float16* w2 = w216 + (size_t)l * F * E;
    const float* g   = ln_g + (size_t)l * E;
    const float* bb  = ln_b + (size_t)l * E;
    const float* bb1 = b1 + (size_t)l * F;
    const float* bb2 = b2 + (size_t)l * E;

    // phi_q/phi_k = elu(h@W)+1 (f16 out only); v = h@Wv (f16 out only)
    launch_gemm<64, false>(h16, wq, nullptr, nullptr, nullptr, phiq16,
                           (int)BS, P, E, 1, 0, 0, 0, 1, stream);
    launch_gemm<64, false>(h16, wk, nullptr, nullptr, nullptr, phik16,
                           (int)BS, P, E, 1, 0, 0, 0, 1, stream);
    launch_gemm<128, false>(h16, wv, nullptr, nullptr, nullptr, s16,
                            (int)BS, E, E, 0, 0, 0, 0, 1, stream);

    // kv[b] = phik[b]^T @ v[b]  (M=P, N=E, K=S, batched, f16 out)
    launch_gemm<128, true>(phik16, s16, nullptr, nullptr, nullptr, kv16,
                           P, E, S, 0,
                           (long long)S * P, (long long)S * E, (long long)P * E, B, stream);

    // z[b] = sum_s phik[b,s,:]
    perf_zreduce<<<dim3(B * P), dim3(256), 0, stream>>>(phik16, zb, S, P);

    // num[b] = phiq[b] @ kv[b]  (fp32 out)
    launch_gemm<128, false>(phiq16, kv16, nullptr, nullptr, tF, nullptr,
                            S, E, P, 0,
                            (long long)S * P, (long long)P * E, (long long)S * E, B, stream);

    // attn16 = num / (phiq.z + eps)
    perf_denscale<<<dim3((unsigned)BS), dim3(64), 0, stream>>>(phiq16, zb, tF, s16, S, P, E);

    // a = attn @ Wo (fp32, overwrites num) ; n16 = layernorm(a)
    launch_gemm<128, false>(s16, wo, nullptr, nullptr, tF, nullptr,
                            (int)BS, E, E, 0, 0, 0, 0, 1, stream);
    perf_layernorm<<<dim3((unsigned)BS), dim3(256), 0, stream>>>(tF, s16, g, bb, E);

    // FFN chunks: m16 = gelu(n@W1+b1); h_next = m@W2 + b2 + resid (fp32+f16)
    for (long long r0 = 0; r0 < BS; r0 += CHUNK) {
      launch_gemm<128, false>(s16 + r0 * E, w1, bb1, nullptr, nullptr, m16,
                              CHUNK, F, E, 2, 0, 0, 0, 1, stream);
      launch_gemm<128, false>(m16, w2, bb2, hF + r0 * E, hFalt + r0 * E, h16alt + r0 * E,
                              CHUNK, E, F, 0, 0, 0, 0, 1, stream);
    }

    { float* t = hF; hF = hFalt; hFalt = t; }
    { _Float16* t = h16; h16 = h16alt; h16alt = t; }
  }

  perf_head<<<dim3(B), dim3(256), 0, stream>>>(hF, Wh1, bh1, Wh2, bh2, out, S, E, H, C);
}
